// SwinTransformerBlock_89593017795201
// MI455X (gfx1250) — matrix operand order repaired and checked by reading, not compiled, not run on hardware
//
#include <hip/hip_runtime.h>
#include <hip/hip_bf16.h>

typedef __attribute__((ext_vector_type(16))) _Float16 v16h;
typedef __attribute__((ext_vector_type(8)))  float    v8f;

#define PIX    73728     // B*H*W = 2*192*192
#define HIMG   192
#define WIMG   192
#define CIMG   192
#define HWIMG  36864
#define NWIN   1152      // 2 * 24 * 24
#define NWIMG  576
#define HEADS  6

// ---------------- CDNA5 async global->LDS DMA (ASYNCcnt-tracked) -------------
// VFLAT encoding: VDST = VGPR holding LDS byte address, VADDR = 64-bit global.
static __device__ __forceinline__ void async_load_b128(const void* gptr, unsigned lds_addr) {
  asm volatile("global_load_async_to_lds_b128 %0, %1, off"
               :: "v"(lds_addr), "v"(gptr) : "memory");
}
static __device__ __forceinline__ void wait_async0() {
  asm volatile("s_wait_asynccnt 0x0" ::: "memory");
}

// ---------------- WMMA fragment loaders (CDNA5 16x16x32 f16 layouts) ----------

// A matrix 16x32 f16: lane%16 = row M; lane/16 selects K-halves.
// element e -> K = (e%8) + (e/8)*16 + (lane/16)*8  => two contiguous 16B chunks.
static __device__ __forceinline__ v16h load_a16(const _Float16* base, int lda, int k0, int lane) {
  const _Float16* p = base + (size_t)(lane & 15) * lda + k0 + ((lane >> 4) << 3);
  v16h a;
  ((float4*)&a)[0] = *(const float4*)(p);
  ((float4*)&a)[1] = *(const float4*)(p + 16);
  return a;
}

// B matrix 32x16 f16 from row-major Wt[N][K]: lane%16 = col N,
// element e -> K = (e%8) + (e/8)*16 + (lane/16)*8 (same as A).
static __device__ __forceinline__ v16h load_b16(const _Float16* Wt, int ldb, int n0, int k0, int lane) {
  const _Float16* p = Wt + (size_t)(n0 + (lane & 15)) * ldb + k0 + ((lane >> 4) << 3);
  v16h b;
  ((float4*)&b)[0] = *(const float4*)(p);
  ((float4*)&b)[1] = *(const float4*)(p + 16);
  return b;
}

static __device__ __forceinline__ v8f wmma_f16(v16h a, v16h b, v8f c) {
  return __builtin_amdgcn_wmma_f32_16x16x32_f16(false, a, false, b, (short)0, c, false, false);
}

// ---------------- weight conversion / padding --------------------------------

__global__ __launch_bounds__(256) void prep_w_kernel(
    const float* __restrict__ qkv_w, const float* __restrict__ proj_w,
    const float* __restrict__ fin_w, const float* __restrict__ fout_w,
    _Float16* __restrict__ WQ, _Float16* __restrict__ WP,
    _Float16* __restrict__ WI, _Float16* __restrict__ WO)
{
  int i = blockIdx.x * 256 + threadIdx.x;          // grid covers 1024*192
  if (i < 576 * 192) WQ[i] = (_Float16)qkv_w[i];
  if (i < 192 * 192) WP[i] = (_Float16)proj_w[i];
  if (i < 1024 * 192) {                            // ffn_in padded 1020 -> 1024 rows
    int nn = i / 192, kk = i % 192;
    WI[i] = (_Float16)(nn < 1020 ? fin_w[nn * 192 + kk] : 0.0f);
  }
  if (i < 192 * 512) {                             // ffn_out padded K 510 -> 512
    int nn = i / 512, kk = i % 512;
    WO[i] = (_Float16)(kk < 510 ? fout_w[nn * 510 + kk] : 0.0f);
  }
}

// ---------------- LayerNorm over channels ------------------------------------

__global__ __launch_bounds__(256) void ln_nchw_kernel(
    const float* __restrict__ x, const float* __restrict__ w,
    const float* __restrict__ b, _Float16* __restrict__ xn)
{
  int p = blockIdx.x * 256 + threadIdx.x;          // p < PIX
  int bb = p / HWIMG, rr = p % HWIMG;
  const float* xp = x + (size_t)bb * CIMG * HWIMG + rr;
  float s = 0.f, s2 = 0.f;
  for (int c = 0; c < CIMG; ++c) {
    float v = xp[(size_t)c * HWIMG];
    s += v; s2 += v * v;
  }
  float mu = s * (1.0f / CIMG);
  float var = s2 * (1.0f / CIMG) - mu * mu;
  float inv = rsqrtf(var + 1e-5f);
  for (int c = 0; c < CIMG; ++c) {
    float v = (xp[(size_t)c * HWIMG] - mu) * inv * w[c] + b[c];
    xn[(size_t)p * CIMG + c] = (_Float16)v;
  }
}

__global__ __launch_bounds__(256) void ln_rowmajor_kernel(
    const float* __restrict__ x2, const float* __restrict__ w,
    const float* __restrict__ b, _Float16* __restrict__ xn)
{
  int p = blockIdx.x * 256 + threadIdx.x;
  const float* xp = x2 + (size_t)p * CIMG;
  float s = 0.f, s2 = 0.f;
  for (int c = 0; c < CIMG; ++c) { float v = xp[c]; s += v; s2 += v * v; }
  float mu = s * (1.0f / CIMG);
  float var = s2 * (1.0f / CIMG) - mu * mu;
  float inv = rsqrtf(var + 1e-5f);
  for (int c = 0; c < CIMG; ++c)
    xn[(size_t)p * CIMG + c] = (_Float16)((xp[c] - mu) * inv * w[c] + b[c]);
}

// ---------------- unified WMMA GEMM with async-DMA A staging -----------------
// out[M=PIX][N] = A[M][K] (f16, row-major) x Wt[N][K] (f16, row-major)
// The 128xK A block per workgroup is DMA'd once into LDS via
// GLOBAL_LOAD_ASYNC_TO_LDS_B128, then all 8 waves stream fragments from LDS.
// epi 0: f16 store to outh (stride ostride)
// epi 1: x2[p][n] = shortcut_NCHW[b,n,y,x] + acc  (proj + residual)
// epi 2: d_out NCHW[b,n,y,x] = x2[p][n] + acc     (ffn_out + residual)
__global__ __launch_bounds__(256) void gemm_f16_kernel(
    const _Float16* __restrict__ A, const _Float16* __restrict__ Wt,
    int K, int N, int epi,
    _Float16* __restrict__ outh, int ostride,
    float* __restrict__ outf, const float* __restrict__ resf)
{
  extern __shared__ _Float16 smem[];               // 128 * K f16
  int tid = threadIdx.x;
  int lane = tid & 31;
  int wv = tid >> 5;
  int mblk = blockIdx.x * 128;

  // ---- async DMA: contiguous 128*K f16 block -> LDS ----
  const _Float16* Ablk = A + (size_t)mblk * K;
  unsigned lds0 = (unsigned)(size_t)smem;
  int nchunks = (128 * K) >> 3;                    // 16B chunks
  for (int i = tid; i < nchunks; i += 256)
    async_load_b128(Ablk + (size_t)i * 8, lds0 + i * 16);
  wait_async0();
  __syncthreads();

  const _Float16* Arow = smem + (size_t)(wv * 16) * K;
  int m0 = mblk + wv * 16;
  int ksteps = K >> 5;
  for (int nt = 0; nt < (N >> 4); ++nt) {
    v8f acc = {0.f, 0.f, 0.f, 0.f, 0.f, 0.f, 0.f, 0.f};
    for (int kk = 0; kk < ksteps; ++kk) {
      v16h a = load_a16(Arow, K, kk << 5, lane);   // ds_load_b128 pairs
      v16h b = load_b16(Wt, K, nt << 4, kk << 5, lane);
      acc = wmma_f16(a, b, acc);
    }
    int n = (nt << 4) + (lane & 15);
    if (epi == 0) {
#pragma unroll
      for (int r = 0; r < 8; ++r) {
        int m = m0 + r + ((lane >> 4) << 3);
        outh[(size_t)m * ostride + n] = (_Float16)acc[r];
      }
    } else if (epi == 1) {
#pragma unroll
      for (int r = 0; r < 8; ++r) {
        int m = m0 + r + ((lane >> 4) << 3);
        int b_ = m / HWIMG, rr = m % HWIMG;
        int y = rr / WIMG, x = rr % WIMG;
        float v = resf[((size_t)(b_ * CIMG + n) * HIMG + y) * WIMG + x] + acc[r];
        outf[(size_t)m * CIMG + n] = v;
      }
    } else {
#pragma unroll
      for (int r = 0; r < 8; ++r) {
        int m = m0 + r + ((lane >> 4) << 3);
        int b_ = m / HWIMG, rr = m % HWIMG;
        int y = rr / WIMG, x = rr % WIMG;
        outf[((size_t)(b_ * CIMG + n) * HIMG + y) * WIMG + x] =
            resf[(size_t)m * CIMG + n] + acc[r];
      }
    }
  }
}

// ---------------- per-window depthwise 3x3 on qkv (with cyclic shift fold) ---
// input qkv1[P][576] in ORIGINAL coords; window pixel (i,j) of shifted image
// maps to orig y=(wy*8+i+4)%192. SAME zero padding at window edges.
// Q/K written token-major [win][head][n][32]; V channel-major [win][head][32][n].
__global__ __launch_bounds__(256) void dwconv_qkv_kernel(
    const _Float16* __restrict__ qkv1, const float* __restrict__ dww,
    _Float16* __restrict__ Q, _Float16* __restrict__ Kk, _Float16* __restrict__ Vt)
{
  size_t idx = (size_t)blockIdx.x * 256 + threadIdx.x;  // NWIN*576*64 threads
  int n = (int)(idx & 63);
  size_t rest = idx >> 6;
  int ch = (int)(rest % 576);
  int win = (int)(rest / 576);
  int bb = win / NWIMG, wi = win % NWIMG;
  int wy = wi / 24, wx = wi % 24;
  int i = n >> 3, j = n & 7;
  const float* wts = dww + ch * 9;
  float acc = 0.f;
#pragma unroll
  for (int di = 0; di < 3; ++di) {
    int ii = i + di - 1;
    if (ii < 0 || ii > 7) continue;
    int y = (wy * 8 + ii + 4) % HIMG;
#pragma unroll
    for (int dj = 0; dj < 3; ++dj) {
      int jj = j + dj - 1;
      if (jj < 0 || jj > 7) continue;
      int x = (wx * 8 + jj + 4) % WIMG;
      float v = (float)qkv1[((size_t)(bb * HWIMG + y * WIMG + x)) * 576 + ch];
      acc += v * wts[di * 3 + dj];
    }
  }
  int t = ch / CIMG;           // 0=q 1=k 2=v
  int cc = ch % CIMG;
  int hh = cc >> 5, c = cc & 31;
  size_t base = ((size_t)(win * HEADS + hh)) * 2048;
  _Float16 hv = (_Float16)acc;
  if (t == 0)      Q [base + n * 32 + c] = hv;
  else if (t == 1) Kk[base + n * 32 + c] = hv;
  else             Vt[base + c * 64 + n] = hv;
}

// ---------------- window attention: one workgroup per (window, head) ---------
__global__ __launch_bounds__(128) void attn_kernel(
    const _Float16* __restrict__ Q, const _Float16* __restrict__ Kk,
    const _Float16* __restrict__ Vt, const float* __restrict__ mask,
    const float* __restrict__ rpb, const float* __restrict__ temp,
    _Float16* __restrict__ aout)
{
  int win = blockIdx.x;
  int h = blockIdx.y;
  __shared__ __align__(16) _Float16 qs[64 * 32];
  __shared__ __align__(16) _Float16 ks[64 * 32];
  __shared__ __align__(16) _Float16 vs[32 * 64];   // V transposed: [c][token]
  __shared__ __align__(16) float    S[64 * 64];
  __shared__ __align__(16) _Float16 Pm[64 * 64];
  int tid = threadIdx.x, lane = tid & 31, wv = tid >> 5;

  // async DMA the three 4KB tiles into LDS
  size_t base = ((size_t)(win * HEADS + h)) * 2048;
  unsigned qso = (unsigned)(size_t)qs;
  unsigned kso = (unsigned)(size_t)ks;
  unsigned vso = (unsigned)(size_t)vs;
  for (int i = tid; i < 256; i += 128) {           // 2048 f16 each = 256 x 16B
    async_load_b128(Q  + base + (size_t)i * 8, qso + i * 16);
    async_load_b128(Kk + base + (size_t)i * 8, kso + i * 16);
    async_load_b128(Vt + base + (size_t)i * 8, vso + i * 16);
  }
  wait_async0();
  __syncthreads();

  // l2norm rows of q (fold temperature) and k
  float tpr = temp[h];
  if (tid < 64) {
    float s = 0.f;
    for (int c = 0; c < 32; ++c) { float v = (float)qs[tid * 32 + c]; s += v * v; }
    float sc = tpr / fmaxf(sqrtf(s), 1e-12f);
    for (int c = 0; c < 32; ++c)
      qs[tid * 32 + c] = (_Float16)((float)qs[tid * 32 + c] * sc);
  } else {
    int r = tid - 64;
    float s = 0.f;
    for (int c = 0; c < 32; ++c) { float v = (float)ks[r * 32 + c]; s += v * v; }
    float sc = 1.0f / fmaxf(sqrtf(s), 1e-12f);
    for (int c = 0; c < 32; ++c)
      ks[r * 32 + c] = (_Float16)((float)ks[r * 32 + c] * sc);
  }
  __syncthreads();

  // S[qn][km] = q @ k^T + bias + mask  (wave wv -> query rows m0..m0+15)
  int m0 = wv * 16;
  const float* mrow = mask + (size_t)(win % NWIMG) * 4096;
  v16h aq = load_a16(qs + m0 * 32, 32, 0, lane);
  for (int nt = 0; nt < 4; ++nt) {
    v16h b = load_b16(ks, 32, nt * 16, 0, lane);
    v8f acc = {0.f, 0.f, 0.f, 0.f, 0.f, 0.f, 0.f, 0.f};
    acc = wmma_f16(aq, b, acc);
    int km = nt * 16 + (lane & 15);
    int i2 = km >> 3, j2 = km & 7;
#pragma unroll
    for (int r = 0; r < 8; ++r) {
      int qn = m0 + r + ((lane >> 4) << 3);
      int i1 = qn >> 3, j1 = qn & 7;
      int rel = (i1 - i2 + 7) * 15 + (j1 - j2 + 7);
      S[qn * 64 + km] = acc[r] + rpb[rel * HEADS + h] + mrow[qn * 64 + km];
    }
  }
  __syncthreads();

  // row softmax -> f16
  if (tid < 64) {
    float mx = -1e30f;
    for (int j = 0; j < 64; ++j) mx = fmaxf(mx, S[tid * 64 + j]);
    float sm = 0.f;
    for (int j = 0; j < 64; ++j) { float e = __expf(S[tid * 64 + j] - mx); S[tid * 64 + j] = e; sm += e; }
    float inv = 1.0f / sm;
    for (int j = 0; j < 64; ++j) Pm[tid * 64 + j] = (_Float16)(S[tid * 64 + j] * inv);
  }
  __syncthreads();

  // O[qn][c] = P @ V ; store with window_reverse + un-shift into aout[p][h*32+c]
  int bwin = win / NWIMG, wi = win % NWIMG;
  int wy = wi / 24, wx = wi % 24;
  for (int nt = 0; nt < 2; ++nt) {
    v8f acc = {0.f, 0.f, 0.f, 0.f, 0.f, 0.f, 0.f, 0.f};
    for (int kk = 0; kk < 2; ++kk) {
      v16h a = load_a16(Pm + m0 * 64, 64, kk * 32, lane);
      v16h b = load_b16(vs, 64, nt * 16, kk * 32, lane);
      acc = wmma_f16(a, b, acc);
    }
    int c = nt * 16 + (lane & 15);
#pragma unroll
    for (int r = 0; r < 8; ++r) {
      int qn = m0 + r + ((lane >> 4) << 3);
      int i = qn >> 3, j = qn & 7;
      int y = (wy * 8 + i + 4) % HIMG;
      int x = (wx * 8 + j + 4) % WIMG;
      aout[((size_t)(bwin * HWIMG + y * WIMG + x)) * CIMG + h * 32 + c] = (_Float16)acc[r];
    }
  }
}

// ---------------- full-image depthwise 3x3 + GELU gate for FFN ---------------
// h1[P][1024] (ch 0..1019 valid). g[p][ch<510] = gelu(conv(ch)) * conv(ch+510).
__global__ __launch_bounds__(256) void dwconv_ffn_gate_kernel(
    const _Float16* __restrict__ h1, const float* __restrict__ dww,
    _Float16* __restrict__ g)
{
  size_t idx = (size_t)blockIdx.x * 256 + threadIdx.x;   // PIX*512 threads
  int ch = (int)(idx & 511);
  int p = (int)(idx >> 9);
  if (ch >= 510) { g[(size_t)p * 512 + ch] = (_Float16)0.0f; return; }
  int bb = p / HWIMG, rr = p % HWIMG;
  int y = rr / WIMG, x = rr % WIMG;
  const float* w1 = dww + ch * 9;
  const float* w2 = dww + (ch + 510) * 9;
  float a = 0.f, bv = 0.f;
#pragma unroll
  for (int di = 0; di < 3; ++di) {
    int yy = y + di - 1;
    if (yy < 0 || yy >= HIMG) continue;
#pragma unroll
    for (int dj = 0; dj < 3; ++dj) {
      int xx = x + dj - 1;
      if (xx < 0 || xx >= WIMG) continue;
      size_t pp = ((size_t)(bb * HWIMG + yy * WIMG + xx)) * 1024;
      a  += (float)h1[pp + ch]       * w1[di * 3 + dj];
      bv += (float)h1[pp + ch + 510] * w2[di * 3 + dj];
    }
  }
  float ge = 0.5f * a * (1.0f + erff(a * 0.70710678f));
  g[(size_t)p * 512 + ch] = (_Float16)(ge * bv);
}

// ---------------- workspace layout -------------------------------------------
static constexpr size_t OFF_WQ   = 0;
static constexpr size_t SZ_WQ    = (size_t)576 * 192 * 2;
static constexpr size_t OFF_WP   = OFF_WQ + SZ_WQ;
static constexpr size_t SZ_WP    = (size_t)192 * 192 * 2;
static constexpr size_t OFF_WI   = OFF_WP + SZ_WP;
static constexpr size_t SZ_WI    = (size_t)1024 * 192 * 2;
static constexpr size_t OFF_WO   = OFF_WI + SZ_WI;
static constexpr size_t SZ_WO    = (size_t)192 * 512 * 2;
static constexpr size_t OFF_XN   = OFF_WO + SZ_WO;
static constexpr size_t SZ_XN    = (size_t)PIX * 192 * 2;
static constexpr size_t OFF_AOUT = OFF_XN + SZ_XN;
static constexpr size_t OFF_X2   = OFF_AOUT + SZ_XN;
static constexpr size_t SZ_X2    = (size_t)PIX * 192 * 4;
static constexpr size_t OFF_R    = OFF_X2 + SZ_X2;
static constexpr size_t OFF_QKV1 = OFF_R;
static constexpr size_t SZ_QKV1  = (size_t)PIX * 576 * 2;
static constexpr size_t SZ_QH    = (size_t)NWIN * HEADS * 2048 * 2;
static constexpr size_t OFF_Q    = OFF_QKV1 + SZ_QKV1;
static constexpr size_t OFF_K    = OFF_Q + SZ_QH;
static constexpr size_t OFF_V    = OFF_K + SZ_QH;
static constexpr size_t OFF_H1   = OFF_R;                 // reuse (qkv dead)
static constexpr size_t SZ_H1    = (size_t)PIX * 1024 * 2;
static constexpr size_t OFF_G    = OFF_H1 + SZ_H1;

extern "C" void kernel_launch(void* const* d_in, const int* in_sizes, int n_in,
                              void* d_out, int out_size, void* d_ws, size_t ws_size,
                              hipStream_t stream) {
  (void)in_sizes; (void)n_in; (void)out_size; (void)ws_size;
  const float* x      = (const float*)d_in[0];
  const float* mask   = (const float*)d_in[1];
  const float* n1w    = (const float*)d_in[2];
  const float* n1b    = (const float*)d_in[3];
  const float* qkv_w  = (const float*)d_in[4];
  const float* qkv_dw = (const float*)d_in[5];
  const float* temp   = (const float*)d_in[6];
  const float* rpb    = (const float*)d_in[7];
  const float* proj_w = (const float*)d_in[8];
  const float* n2w    = (const float*)d_in[9];
  const float* n2b    = (const float*)d_in[10];
  const float* fin_w  = (const float*)d_in[11];
  const float* fdw_w  = (const float*)d_in[12];
  const float* fout_w = (const float*)d_in[13];
  float* out = (float*)d_out;
  char* ws = (char*)d_ws;

  _Float16* WQ   = (_Float16*)(ws + OFF_WQ);
  _Float16* WP   = (_Float16*)(ws + OFF_WP);
  _Float16* WI   = (_Float16*)(ws + OFF_WI);
  _Float16* WO   = (_Float16*)(ws + OFF_WO);
  _Float16* XN   = (_Float16*)(ws + OFF_XN);
  _Float16* AOUT = (_Float16*)(ws + OFF_AOUT);
  float*    X2   = (float*)   (ws + OFF_X2);
  _Float16* QKV1 = (_Float16*)(ws + OFF_QKV1);
  _Float16* Qb   = (_Float16*)(ws + OFF_Q);
  _Float16* Kb   = (_Float16*)(ws + OFF_K);
  _Float16* Vb   = (_Float16*)(ws + OFF_V);
  _Float16* H1   = (_Float16*)(ws + OFF_H1);
  _Float16* G    = (_Float16*)(ws + OFF_G);

  prep_w_kernel<<<768, 256, 0, stream>>>(qkv_w, proj_w, fin_w, fout_w, WQ, WP, WI, WO);
  ln_nchw_kernel<<<PIX / 256, 256, 0, stream>>>(x, n1w, n1b, XN);
  // qkv 1x1: [PIX,192] x [576,192]^T -> f16 [PIX,576]   (LDS A: 128*192*2 = 48KB)
  gemm_f16_kernel<<<PIX / 128, 256, 128 * 192 * 2, stream>>>(XN, WQ, 192, 576, 0, QKV1, 576, nullptr, nullptr);
  dwconv_qkv_kernel<<<(NWIN * 576 * 64) / 256, 256, 0, stream>>>(QKV1, qkv_dw, Qb, Kb, Vb);
  attn_kernel<<<dim3(NWIN, HEADS), 128, 0, stream>>>(Qb, Kb, Vb, mask, rpb, temp, AOUT);
  // proj + shortcut -> x2 f32 [PIX,192]
  gemm_f16_kernel<<<PIX / 128, 256, 128 * 192 * 2, stream>>>(AOUT, WP, 192, 192, 1, nullptr, 0, X2, x);
  ln_rowmajor_kernel<<<PIX / 256, 256, 0, stream>>>(X2, n2w, n2b, XN);
  // ffn_in: [PIX,192] x [1024,192]^T -> f16 [PIX,1024]
  gemm_f16_kernel<<<PIX / 128, 256, 128 * 192 * 2, stream>>>(XN, WI, 192, 1024, 0, H1, 1024, nullptr, nullptr);
  dwconv_ffn_gate_kernel<<<(PIX * 512) / 256, 256, 0, stream>>>(H1, fdw_w, G);
  // ffn_out + residual -> d_out NCHW f32  (LDS A: 128*512*2 = 128KB, within 320KB WGP)
  gemm_f16_kernel<<<PIX / 128, 256, 128 * 512 * 2, stream>>>(G, WO, 512, 192, 2, nullptr, 0, out, X2);
}
